// GATMeshNetwork_20590073217162
// MI455X (gfx1250) — matrix operand order repaired and checked by reading, not compiled, not run on hardware
//
#include <hip/hip_runtime.h>
#include <hip/hip_bf16.h>

// ---------------- CDNA5 WMMA types ----------------
typedef __attribute__((ext_vector_type(16))) _Float16 v16h;
typedef __attribute__((ext_vector_type(8)))  _Float16 f16x8;
typedef __attribute__((ext_vector_type(8)))  float    v8f;

#define NPc 100000LL
#define NGc 2000LL
#define EPc 800000LL
#define NMc 2000LL
#define EMc 32000LL

// ---------------- helpers ----------------
__device__ __forceinline__ float leaky(float v, float s) {
  return v >= 0.f ? v : s * v;
}

// float atomic max via sign-split int punning (correct total order for mixed signs)
__device__ __forceinline__ void atomicMaxF(float* addr, float v) {
  if (v >= 0.f) atomicMax((int*)addr, __float_as_int(v));
  else          atomicMin((unsigned int*)addr, __float_as_uint(v));
}

// ---------------- generic fill ----------------
__global__ void fill_k(float* p, float v, long long n) {
  long long i = (long long)blockIdx.x * blockDim.x + threadIdx.x;
  if (i < n) p[i] = v;
}

// ---------------- f32 -> f16 staging (zero-padded K) --------------------------
// Ah[row, k] = (k < K) ? A[row, k] : 0      (Ah is [N, Kp] row-major)
__global__ void cvt_a_k(const float* __restrict__ A, _Float16* __restrict__ Ah,
                        long long N, int K, int Kp) {
  long long i = (long long)blockIdx.x * blockDim.x + threadIdx.x;
  if (i >= N * Kp) return;
  long long row = i / Kp; int k = (int)(i % Kp);
  Ah[i] = (k < K) ? (_Float16)A[row * K + k] : (_Float16)0.f;
}
// Bt[m, k] = (k < K) ? B[k, m] : 0          (Bt is [M, Kp] row-major == B^T)
__global__ void cvt_bt_k(const float* __restrict__ B, _Float16* __restrict__ Bt,
                         int K, int M, int Kp) {
  long long i = (long long)blockIdx.x * blockDim.x + threadIdx.x;
  if (i >= (long long)M * Kp) return;
  int m = (int)(i / Kp); int k = (int)(i % Kp);
  Bt[i] = (k < K) ? (_Float16)B[(long long)k * M + m] : (_Float16)0.f;
}

// ---------------- WMMA GEMM core: C[N,M] = Ah[N,Kp] @ Bt[M,Kp]^T --------------
// N, M multiples of 16; Kp multiple of 32. 4 waves/block, one 16x16 tile each.
// Inner loop: 4x global_load_b128 + 1x v_wmma, no predication (EXEC all-ones).
__global__ void gemm_wmma_k(const _Float16* __restrict__ Ah,
                            const _Float16* __restrict__ Bt,
                            float* __restrict__ C, int N, int Kp, int M) {
  const int lane = threadIdx.x & 31;
  const int wv = threadIdx.x >> 5;
  const int hs = lane >> 4;          // half-select
  const int lr = lane & 15;
  const int rowTile = blockIdx.x * 4 + wv;
  if (rowTile * 16 >= N) return;     // wave-uniform
  const int row0 = rowTile * 16;
  const int col0 = blockIdx.y * 16;

  const _Float16* ap = Ah + (long long)(row0 + lr) * Kp + (hs << 3);
  const _Float16* bp = Bt + (long long)(col0 + lr) * Kp + (hs << 3);

  v8f acc = {};
  for (int k0 = 0; k0 < Kp; k0 += 32) {
    union { v16h v; f16x8 h[2]; } af, bf;
    af.h[0] = *(const f16x8*)(ap + k0);        // K = k0 + 8*hs + [0..7]
    af.h[1] = *(const f16x8*)(ap + k0 + 16);   // K = k0 + 16 + 8*hs + [0..7]
    bf.h[0] = *(const f16x8*)(bp + k0);        // K = k0 + 8*hs + [0..7]
    bf.h[1] = *(const f16x8*)(bp + k0 + 16);   // K = k0 + 16 + 8*hs + [0..7]
    acc = __builtin_amdgcn_wmma_f32_16x16x32_f16(false, af.v, false, bf.v,
                                                 (short)0, acc, false, false);
  }
  float* cp = C + (long long)(row0 + (hs << 3)) * M + col0 + lr;
#pragma unroll
  for (int r = 0; r < 8; ++r) cp[(long long)r * M] = acc[r];
}

// ---------------- per-node attention dots: el/er [N,H] ----------------
__global__ void attn_k(const float* __restrict__ h, const float* __restrict__ al,
                       const float* __restrict__ ar, float* el, float* er,
                       long long N, int H, int D) {
  long long i = (long long)blockIdx.x * blockDim.x + threadIdx.x;
  if (i >= N * H) return;
  long long n = i / H; int hh = (int)(i % H);
  const float* hp = h + n * (long long)H * D + (long long)hh * D;
  float sl = 0.f, sr = 0.f;
  for (int d = 0; d < D; ++d) { sl += hp[d] * al[hh * D + d]; sr += hp[d] * ar[hh * D + d]; }
  el[i] = sl; er[i] = sr;
}

// ---------------- edge pass 1: e = leaky(el[src]+er[dst],0.2); segment max ----
__global__ void edge_max_k(const float* __restrict__ el, const float* __restrict__ er,
                           const int* __restrict__ src, const int* __restrict__ dst,
                           float* e, float* m, long long E, int H) {
  long long i = (long long)blockIdx.x * blockDim.x + threadIdx.x;
  if (i >= E * H) return;
  long long eid = i / H; int hh = (int)(i % H);
  float v = el[(long long)src[eid] * H + hh] + er[(long long)dst[eid] * H + hh];
  v = leaky(v, 0.2f);
  e[i] = v;
  atomicMaxF(&m[(long long)dst[eid] * H + hh], v);
}

// ---------------- edge pass 2: ee = exp(e - m[dst]); segment sum --------------
__global__ void edge_expsum_k(float* e, const int* __restrict__ dst,
                              const float* __restrict__ m, float* s,
                              long long E, int H) {
  long long i = (long long)blockIdx.x * blockDim.x + threadIdx.x;
  if (i >= E * H) return;
  long long eid = i / H; int hh = (int)(i % H);
  float ee = expf(e[i] - m[(long long)dst[eid] * H + hh]);
  e[i] = ee;
  atomicAdd(&s[(long long)dst[eid] * H + hh], ee);
}

// ---------------- edge pass 3: out[dst] += h[src] * (ee / max(s,1e-9)) --------
__global__ void edge_scatter_k(const float* __restrict__ e, const int* __restrict__ src,
                               const int* __restrict__ dst, const float* __restrict__ h,
                               const float* __restrict__ s, float* out,
                               long long E, int H, int D) {
  long long i = (long long)blockIdx.x * blockDim.x + threadIdx.x;
  if (i >= E * (long long)H * D) return;
  int d = (int)(i % D); long long t = i / D;
  int hh = (int)(t % H); long long eid = t / H;
  long long ds = dst[eid], sr = src[eid];
  float a = e[eid * H + hh] / fmaxf(s[ds * H + hh], 1e-9f);
  atomicAdd(&out[ds * (long long)H * D + (long long)hh * D + d],
            h[sr * (long long)H * D + (long long)hh * D + d] * a);
}

// ---------------- fused LeakyReLU(0.01) + GraphNorm (block per channel) -------
__global__ void leaky_gnorm_k(const float* __restrict__ x, const float* __restrict__ g,
                              const float* __restrict__ b, const float* __restrict__ al,
                              float* y, long long Nn, int C, float slope) {
  const int c = blockIdx.x;
  const int tid = threadIdx.x;
  __shared__ float red[256];
  __shared__ float s_mu, s_inv;
  float acc = 0.f;
  for (long long r = tid; r < Nn; r += 256) acc += leaky(x[r * C + c], slope);
  red[tid] = acc; __syncthreads();
  for (int st = 128; st > 0; st >>= 1) { if (tid < st) red[tid] += red[tid + st]; __syncthreads(); }
  if (tid == 0) s_mu = red[0] / (float)Nn;
  __syncthreads();
  const float mu = s_mu, a = al[c];
  acc = 0.f;
  for (long long r = tid; r < Nn; r += 256) {
    float xc = leaky(x[r * C + c], slope) - a * mu; acc += xc * xc;
  }
  red[tid] = acc; __syncthreads();
  for (int st = 128; st > 0; st >>= 1) { if (tid < st) red[tid] += red[tid + st]; __syncthreads(); }
  if (tid == 0) s_inv = rsqrtf(red[0] / (float)Nn + 1e-5f);
  __syncthreads();
  const float inv = s_inv, gg = g[c], bb = b[c];
  for (long long r = tid; r < Nn; r += 256)
    y[r * C + c] = gg * (leaky(x[r * C + c], slope) - a * mu) * inv + bb;
}

// ---------------- mean over heads: [N,H,D] -> [N,D] ---------------------------
__global__ void head_mean_k(const float* __restrict__ x, float* y,
                            long long N, int H, int D) {
  long long i = (long long)blockIdx.x * blockDim.x + threadIdx.x;
  if (i >= N * D) return;
  long long n = i / D; int d = (int)(i % D);
  float s = 0.f;
  for (int hh = 0; hh < H; ++hh) s += x[n * (long long)H * D + (long long)hh * D + d];
  y[i] = s / (float)H;
}

// ---------------- patch-graph readout (segment mean, 32 feats) ----------------
__global__ void readout_acc_k(const float* __restrict__ h, const int* __restrict__ gid,
                              float* rsum, float* cnt, long long NPn) {
  long long i = (long long)blockIdx.x * blockDim.x + threadIdx.x;
  if (i >= NPn * 32) return;
  long long n = i / 32; int k = (int)(i % 32);
  int gidx = gid[n];
  atomicAdd(&rsum[(long long)gidx * 32 + k], h[i]);
  if (k == 0) atomicAdd(&cnt[gidx], 1.0f);
}
__global__ void readout_div_k(float* rsum, const float* __restrict__ cnt, long long NGn) {
  long long i = (long long)blockIdx.x * blockDim.x + threadIdx.x;
  if (i >= NGn * 32) return;
  rsum[i] /= fmaxf(cnt[i / 32], 1.0f);
}

// ---------------- final: colmean over NM rows -> @ mcls[32,15] ----------------
__global__ void final_k(const float* __restrict__ m, const float* __restrict__ mcls,
                        float* out, int NMn) {
  __shared__ float sh[32];
  int t = threadIdx.x;
  if (t < 32) {
    float s = 0.f;
    for (int r = 0; r < NMn; ++r) s += m[(long long)r * 32 + t];
    sh[t] = s / (float)NMn;
  }
  __syncthreads();
  if (t < 15) {
    float s = 0.f;
    for (int k = 0; k < 32; ++k) s += sh[k] * mcls[k * 15 + t];
    out[t] = s;
  }
}

// ---------------- host-side helpers ----------------
static inline unsigned gdiv(long long n) { return (unsigned)((n + 255) / 256); }

// C[N,M] = A[N,K] @ B[K,M]; stages A,B into padded f16 (Ah, Bt) first.
static void gemm(const float* A, const float* B, float* C, int N, int K, int M,
                 _Float16* Ah, _Float16* Bt, hipStream_t st) {
  const int Kp = (K + 31) & ~31;
  cvt_a_k<<<gdiv((long long)N * Kp), 256, 0, st>>>(A, Ah, N, K, Kp);
  cvt_bt_k<<<gdiv((long long)M * Kp), 256, 0, st>>>(B, Bt, K, M, Kp);
  dim3 grid((unsigned)((N / 16 + 3) / 4), (unsigned)(M / 16));
  gemm_wmma_k<<<grid, 128, 0, st>>>(Ah, Bt, C, N, Kp, M);
}

static void gat_edges(const float* h, const float* alw, const float* arw,
                      const int* src, const int* dst,
                      float* el, float* er, float* mx, float* sm, float* ebuf,
                      float* out, long long N, long long E, int H, int D,
                      hipStream_t st) {
  long long NH = N * H, ND = N * (long long)H * D, EH = E * H, EHD = EH * D;
  attn_k<<<gdiv(NH), 256, 0, st>>>(h, alw, arw, el, er, N, H, D);
  fill_k<<<gdiv(NH), 256, 0, st>>>(mx, -__builtin_huge_valf(), NH);
  fill_k<<<gdiv(NH), 256, 0, st>>>(sm, 0.f, NH);
  fill_k<<<gdiv(ND), 256, 0, st>>>(out, 0.f, ND);
  edge_max_k<<<gdiv(EH), 256, 0, st>>>(el, er, src, dst, ebuf, mx, E, H);
  edge_expsum_k<<<gdiv(EH), 256, 0, st>>>(ebuf, dst, mx, sm, E, H);
  edge_scatter_k<<<gdiv(EHD), 256, 0, st>>>(ebuf, src, dst, h, sm, out, E, H, D);
}

extern "C" void kernel_launch(void* const* d_in, const int* in_sizes, int n_in,
                              void* d_out, int out_size, void* d_ws, size_t ws_size,
                              hipStream_t stream) {
  (void)in_sizes; (void)n_in; (void)out_size; (void)ws_size;
  const float* node_feats = (const float*)d_in[0];
  const int*   psrc = (const int*)d_in[1];
  const int*   pdst = (const int*)d_in[2];
  const int*   pgid = (const int*)d_in[3];
  const int*   msrc = (const int*)d_in[4];
  const int*   mdst = (const int*)d_in[5];
  const float *pW1 = (const float*)d_in[6],  *pal1 = (const float*)d_in[7],  *par1 = (const float*)d_in[8];
  const float *pg1g = (const float*)d_in[9], *pg1b = (const float*)d_in[10], *pg1a = (const float*)d_in[11];
  const float *pW2 = (const float*)d_in[12], *pal2 = (const float*)d_in[13], *par2 = (const float*)d_in[14];
  const float *pg2g = (const float*)d_in[15],*pg2b = (const float*)d_in[16], *pg2a = (const float*)d_in[17];
  const float *pW3 = (const float*)d_in[18], *pal3 = (const float*)d_in[19], *par3 = (const float*)d_in[20];
  const float *pcls = (const float*)d_in[21];
  const float *rng = (const float*)d_in[22], *rnb = (const float*)d_in[23], *rna = (const float*)d_in[24];
  const float *mW1 = (const float*)d_in[25], *mal1 = (const float*)d_in[26], *mar1 = (const float*)d_in[27];
  const float *mg1g = (const float*)d_in[28],*mg1b = (const float*)d_in[29], *mg1a = (const float*)d_in[30];
  const float *mW2 = (const float*)d_in[31], *mal2 = (const float*)d_in[32], *mar2 = (const float*)d_in[33];
  const float *mg2g = (const float*)d_in[34],*mg2b = (const float*)d_in[35], *mg2a = (const float*)d_in[36];
  const float *mW3 = (const float*)d_in[37], *mal3 = (const float*)d_in[38], *mar3 = (const float*)d_in[39];
  const float *mcls = (const float*)d_in[40];

  float* out = (float*)d_out;        // [0..14] mesh_out, [15..64014] ro
  float* ro_out = out + 15;

  // workspace layout; f16 staging first (16B-aligned sizes), then f32 buffers
  _Float16* Ah = (_Float16*)d_ws;       // NP*256 halfs (max A tile, 51.2MB)
  _Float16* Bt = Ah + NPc * 256;        // 256*256 halfs (max B^T tile)
  float* bufA = (float*)(Bt + 256 * 256); // NP*256 f32
  float* bufB = bufA + NPc * 256;       // NP*256
  float* el   = bufB + NPc * 256;       // NP*4
  float* er   = el + NPc * 4;
  float* mx   = er + NPc * 4;
  float* sm   = mx + NPc * 4;
  float* ebuf = sm + NPc * 4;           // EP*4
  float* rsum = ebuf + EPc * 4;         // NG*32
  float* cnt  = rsum + NGc * 32;        // NG
  float* rtmp = cnt + NGc;              // NG*32
  float* mA   = rtmp + NGc * 32;        // NM*256
  float* mB   = mA + NMc * 256;         // NM*256

  // ===== patch stage =====
  // GAT1: [NP,5]@[5,256], H=4, D=64
  gemm(node_feats, pW1, bufA, (int)NPc, 5, 256, Ah, Bt, stream);
  gat_edges(bufA, pal1, par1, psrc, pdst, el, er, mx, sm, ebuf, bufB, NPc, EPc, 4, 64, stream);
  leaky_gnorm_k<<<256, 256, 0, stream>>>(bufB, pg1g, pg1b, pg1a, bufA, NPc, 256, 0.01f);
  // GAT2: [NP,256]@[256,192], H=3, D=64
  gemm(bufA, pW2, bufB, (int)NPc, 256, 192, Ah, Bt, stream);
  gat_edges(bufB, pal2, par2, psrc, pdst, el, er, mx, sm, ebuf, bufA, NPc, EPc, 3, 64, stream);
  leaky_gnorm_k<<<192, 256, 0, stream>>>(bufA, pg2g, pg2b, pg2a, bufB, NPc, 192, 0.01f);
  // GAT3: [NP,192]@[192,64], H=2, D=32, then head-mean -> [NP,32]
  gemm(bufB, pW3, bufA, (int)NPc, 192, 64, Ah, Bt, stream);
  gat_edges(bufA, pal3, par3, psrc, pdst, el, er, mx, sm, ebuf, bufB, NPc, EPc, 2, 32, stream);
  head_mean_k<<<gdiv(NPc * 32), 256, 0, stream>>>(bufB, bufA, NPc, 2, 32);
  // readout -> ro = mean_nodes @ pcls, leaky+gnorm -> ro_out (d_out+15)
  fill_k<<<gdiv(NGc * 32), 256, 0, stream>>>(rsum, 0.f, NGc * 32);
  fill_k<<<gdiv(NGc), 256, 0, stream>>>(cnt, 0.f, NGc);
  readout_acc_k<<<gdiv(NPc * 32), 256, 0, stream>>>(bufA, pgid, rsum, cnt, NPc);
  readout_div_k<<<gdiv(NGc * 32), 256, 0, stream>>>(rsum, cnt, NGc);
  gemm(rsum, pcls, rtmp, (int)NGc, 32, 32, Ah, Bt, stream);
  leaky_gnorm_k<<<32, 256, 0, stream>>>(rtmp, rng, rnb, rna, ro_out, NGc, 32, 0.01f);

  // ===== mesh stage (input = ro_out [2000,32]) =====
  gemm(ro_out, mW1, mA, (int)NMc, 32, 256, Ah, Bt, stream);
  gat_edges(mA, mal1, mar1, msrc, mdst, el, er, mx, sm, ebuf, mB, NMc, EMc, 4, 64, stream);
  leaky_gnorm_k<<<256, 256, 0, stream>>>(mB, mg1g, mg1b, mg1a, mA, NMc, 256, 0.01f);
  gemm(mA, mW2, mB, (int)NMc, 256, 96, Ah, Bt, stream);
  gat_edges(mB, mal2, mar2, msrc, mdst, el, er, mx, sm, ebuf, mA, NMc, EMc, 3, 32, stream);
  leaky_gnorm_k<<<96, 256, 0, stream>>>(mA, mg2g, mg2b, mg2a, mB, NMc, 96, 0.01f);
  gemm(mB, mW3, mA, (int)NMc, 96, 64, Ah, Bt, stream);
  gat_edges(mA, mal3, mar3, msrc, mdst, el, er, mx, sm, ebuf, mB, NMc, EMc, 2, 32, stream);
  head_mean_k<<<gdiv(NMc * 32), 256, 0, stream>>>(mB, mA, NMc, 2, 32);
  final_k<<<1, 32, 0, stream>>>(mA, mcls, out, (int)NMc);
}
